// TernaryDense_61753039782097
// MI455X (gfx1250) — compile-verified, matrix operand order checked by reading, not run on hardware
//
#include <hip/hip_runtime.h>

typedef __attribute__((ext_vector_type(16))) _Float16 v16h;
typedef __attribute__((ext_vector_type(8)))  _Float16 v8h;
typedef __attribute__((ext_vector_type(8)))  float    v8f;

#define LDSS 40      // padded K-stride in halfs (80 bytes, 16B aligned)

#if __has_builtin(__builtin_amdgcn_global_load_async_to_lds_b128)
#define HAVE_ASYNC_LDS 1
#else
#define HAVE_ASYNC_LDS 0
#endif

// Match the builtin's parameter types exactly (from the compiler diagnostic):
//   arg0: AS(1) pointer to int __vector(16B)   (global source)
//   arg1: AS(3) pointer to int __vector(16B)   (LDS destination)
typedef int v4i __attribute__((vector_size(16)));
typedef __attribute__((address_space(1))) v4i* as1_v4i;
typedef __attribute__((address_space(3))) v4i* as3_v4i;

static __device__ __forceinline__ as1_v4i as_global16(const void* p) {
    return (as1_v4i)(unsigned long long)p;
}
static __device__ __forceinline__ as3_v4i as_lds16(const void* p) {
    // generic LDS pointer carries the LDS byte offset in its low 32 bits
    return (as3_v4i)(unsigned long long)(unsigned int)(unsigned long long)p;
}

static __device__ __forceinline__ void wait_async_all() {
#if __has_builtin(__builtin_amdgcn_s_wait_asynccnt)
    __builtin_amdgcn_s_wait_asynccnt(0);
#else
    asm volatile("s_wait_asynccnt 0x0" ::: "memory");
#endif
}

// ---------------- Pass 1: A f32 -> f16 ----------------
__global__ __launch_bounds__(256)
void prep_a_f16(const float* __restrict__ A, _Float16* __restrict__ Ah, size_t n) {
    size_t i = ((size_t)blockIdx.x * blockDim.x + threadIdx.x) * 8;
    if (i + 8 > n) return;
    float4 f0 = *(const float4*)(A + i);
    float4 f1 = *(const float4*)(A + i + 4);
    v8h h;
    h[0] = (_Float16)f0.x; h[1] = (_Float16)f0.y; h[2] = (_Float16)f0.z; h[3] = (_Float16)f0.w;
    h[4] = (_Float16)f1.x; h[5] = (_Float16)f1.y; h[6] = (_Float16)f1.z; h[7] = (_Float16)f1.w;
    *(v8h*)(Ah + i) = h;
}

// ------- Pass 2: B f32 [K][N] -> ternary f16, transposed to [N][K] -------
#define TB 64
__global__ __launch_bounds__(256)
void prep_b_tern_t(const float* __restrict__ B, _Float16* __restrict__ Bt, int K, int N) {
    __shared__ _Float16 t[TB][TB + 8];
    const int k0 = blockIdx.y * TB, n0 = blockIdx.x * TB;
    const int tid = threadIdx.x;

    const int kk = tid >> 2;
    const int nn = (tid & 3) * 16;
    const float4* src = (const float4*)(B + (size_t)(k0 + kk) * N + n0 + nn);
    float4 v[4];
    #pragma unroll
    for (int i = 0; i < 4; ++i) v[i] = src[i];
    const float* f = (const float*)v;
    #pragma unroll
    for (int i = 0; i < 16; ++i) {
        float wv = f[i];
        float aw = fabsf(wv);
        float tv = (aw >= 0.5f && aw < 1.5f) ? (wv > 0.0f ? 1.0f : -1.0f) : 0.0f;
        t[kk][nn + i] = (_Float16)tv;
    }
    __syncthreads();

    const int nr = tid >> 2;
    const int kc = (tid & 3) * 16;
    v8h o0, o1;
    #pragma unroll
    for (int i = 0; i < 8; ++i) { o0[i] = t[kc + i][nr]; o1[i] = t[kc + 8 + i][nr]; }
    _Float16* dst = Bt + (size_t)(n0 + nr) * K + k0 + kc;
    *(v8h*)dst       = o0;
    *(v8h*)(dst + 8) = o1;
}

// ---------------- Pass 3: f16 GEMM, C = Ah[M,K] x Bt[N,K]^T ----------------
#define GBM 256
#define GBN 128
#define GBK 32

__global__ __launch_bounds__(256)
void gemm_f16_wmma(const _Float16* __restrict__ Ah,   // [M][K] f16
                   const _Float16* __restrict__ Bt,   // [N][K] f16 (ternary)
                   float* __restrict__ C,             // [M][N] f32
                   int M, int N, int K)
{
    __shared__ _Float16 As[2][GBM * LDSS];
    __shared__ _Float16 Bs[2][GBN * LDSS];

    const int tid  = threadIdx.x;
    const int lane = tid & 31;
    const int w    = tid >> 5;
    const int waveM = w >> 1;        // 0..3 -> 64-row slab
    const int waveN = w & 1;         // 0..1 -> 64-col slab
    const int lm   = lane & 15;
    const int r0   = (lane >> 4) << 3;   // K-run select per 16-bit WMMA layout

    const int bx = blockIdx.x, by = blockIdx.y;

    // A staging: one 32-half row segment per thread (4 x b128)
    const _Float16* Ag = Ah + (size_t)(by * GBM + tid) * K;
    _Float16* AsRow0 = &As[0][tid * LDSS];
    _Float16* AsRow1 = &As[1][tid * LDSS];
    // B staging: 2 threads per row, 16 halfs each (2 x b128)
    const int bn = tid >> 1;
    const int bk = (tid & 1) * 16;
    const _Float16* Bg = Bt + (size_t)(bx * GBN + bn) * K + bk;
    _Float16* BsRow0 = &Bs[0][bn * LDSS + bk];
    _Float16* BsRow1 = &Bs[1][bn * LDSS + bk];

    // Stage tile kt of A and B into LDS buffer `buf`.
    auto stage = [&](int kt, int buf) {
        const _Float16* ga = Ag + (size_t)kt * GBK;
        const _Float16* gb = Bg + (size_t)kt * GBK;
        _Float16* laA = buf ? AsRow1 : AsRow0;
        _Float16* laB = buf ? BsRow1 : BsRow0;
#if HAVE_ASYNC_LDS
        #pragma unroll
        for (int i = 0; i < 4; ++i)
            __builtin_amdgcn_global_load_async_to_lds_b128(
                as_global16(ga + i * 8), as_lds16(laA + i * 8), 0, 0);
        #pragma unroll
        for (int i = 0; i < 2; ++i)
            __builtin_amdgcn_global_load_async_to_lds_b128(
                as_global16(gb + i * 8), as_lds16(laB + i * 8), 0, 0);
#else
        v8h a0 = *(const v8h*)(ga);      v8h a1 = *(const v8h*)(ga + 8);
        v8h a2 = *(const v8h*)(ga + 16); v8h a3 = *(const v8h*)(ga + 24);
        v8h b0 = *(const v8h*)(gb);      v8h b1 = *(const v8h*)(gb + 8);
        *(v8h*)(laA)      = a0; *(v8h*)(laA + 8)  = a1;
        *(v8h*)(laA + 16) = a2; *(v8h*)(laA + 24) = a3;
        *(v8h*)(laB)      = b0; *(v8h*)(laB + 8)  = b1;
#endif
    };

    v8f acc[4][4] = {};

    const int ktiles = K / GBK;
    stage(0, 0);
    wait_async_all();
    __syncthreads();

    for (int kt = 0; kt < ktiles; ++kt) {
        const int cur = kt & 1;

        if (kt + 1 < ktiles) stage(kt + 1, cur ^ 1);   // async: overlaps with WMMA below
        if (kt + 2 < ktiles) {
            __builtin_prefetch(Ag + (size_t)(kt + 2) * GBK, 0, 1);
            __builtin_prefetch(Bg + (size_t)(kt + 2) * GBK, 0, 1);
        }

        v16h af[4], bf[4];
        #pragma unroll
        for (int mi = 0; mi < 4; ++mi) {
            const _Float16* p = &As[cur][(waveM * 64 + mi * 16 + lm) * LDSS];
            v8h lo = *(const v8h*)(p + r0);
            v8h hi = *(const v8h*)(p + r0 + 16);
            #pragma unroll
            for (int i = 0; i < 8; ++i) { af[mi][i] = lo[i]; af[mi][i + 8] = hi[i]; }
        }
        #pragma unroll
        for (int ni = 0; ni < 4; ++ni) {
            const _Float16* p = &Bs[cur][(waveN * 64 + ni * 16 + lm) * LDSS];
            v8h lo = *(const v8h*)(p + r0);
            v8h hi = *(const v8h*)(p + r0 + 16);
            #pragma unroll
            for (int i = 0; i < 8; ++i) { bf[ni][i] = lo[i]; bf[ni][i + 8] = hi[i]; }
        }

        #pragma unroll
        for (int mi = 0; mi < 4; ++mi)
            #pragma unroll
            for (int ni = 0; ni < 4; ++ni)
                acc[mi][ni] = __builtin_amdgcn_wmma_f32_16x16x32_f16(
                    false, af[mi], false, bf[ni], (short)0, acc[mi][ni], false, false);

#if HAVE_ASYNC_LDS
        wait_async_all();
#endif
        __syncthreads();
    }

    const int rowBase = by * GBM + waveM * 64 + ((lane >> 4) << 3);
    const int colBase = bx * GBN + waveN * 64 + lm;
    #pragma unroll
    for (int mi = 0; mi < 4; ++mi)
        #pragma unroll
        for (int ni = 0; ni < 4; ++ni)
            #pragma unroll
            for (int r = 0; r < 8; ++r)
                __builtin_nontemporal_store(
                    acc[mi][ni][r],
                    &C[(size_t)(rowBase + mi * 16 + r) * N + colBase + ni * 16]);
}

// ---------------- Fallback: fused single-kernel path ----------------
#define BM 128
#define BN 128
#define BK 32

__global__ __launch_bounds__(256)
void ternary_gemm_fused(const float* __restrict__ A, const float* __restrict__ B,
                        float* __restrict__ C, int M, int N, int K)
{
    __shared__ _Float16 As[2][BM * LDSS];
    __shared__ _Float16 Bs[2][BN * LDSS];

    const int tid  = threadIdx.x;
    const int lane = tid & 31;
    const int w    = tid >> 5;
    const int waveM = w >> 1;
    const int waveN = w & 1;
    const int lm   = lane & 15;
    const int r0   = (lane >> 4) << 3;
    const int bx = blockIdx.x, by = blockIdx.y;

    const int arow = tid >> 1;
    const int acol = (tid & 1) * 16;
    const float4* Ag = (const float4*)(A + (size_t)(by * BM + arow) * K + acol);
    const int brow = tid >> 3;
    const int bcol = (tid & 7) * 16;
    const float4* Bg = (const float4*)(B + (size_t)brow * N + bx * BN + bcol);

    float4 aReg[4], bReg[4];
    auto loadGlobal = [&](int kt) {
        const float4* pa = Ag + (size_t)kt * (BK / 4);
        #pragma unroll
        for (int i = 0; i < 4; ++i) aReg[i] = pa[i];
        const float4* pb = Bg + (size_t)kt * BK * (N / 4);
        #pragma unroll
        for (int i = 0; i < 4; ++i) bReg[i] = pb[i];
    };
    auto stageToLds = [&](int buf) {
        const float* af = (const float*)aReg;
        v8h h0, h1;
        #pragma unroll
        for (int i = 0; i < 8; ++i) { h0[i] = (_Float16)af[i]; h1[i] = (_Float16)af[i + 8]; }
        *(v8h*)&As[buf][arow * LDSS + acol]     = h0;
        *(v8h*)&As[buf][arow * LDSS + acol + 8] = h1;
        const float* bf = (const float*)bReg;
        #pragma unroll
        for (int i = 0; i < 16; ++i) {
            float wv = bf[i];
            float aw = fabsf(wv);
            float t  = (aw >= 0.5f && aw < 1.5f) ? (wv > 0.0f ? 1.0f : -1.0f) : 0.0f;
            Bs[buf][(bcol + i) * LDSS + brow] = (_Float16)t;
        }
    };

    v8f acc[2][4] = {};
    const int ktiles = K / BK;
    loadGlobal(0);
    stageToLds(0);
    __syncthreads();

    for (int kt = 0; kt < ktiles; ++kt) {
        const int cur = kt & 1;
        if (kt + 1 < ktiles) loadGlobal(kt + 1);

        v16h afrag[2];
        #pragma unroll
        for (int mi = 0; mi < 2; ++mi) {
            const _Float16* p = &As[cur][(waveM * 32 + mi * 16 + lm) * LDSS];
            v8h lo = *(const v8h*)(p + r0);
            v8h hi = *(const v8h*)(p + r0 + 16);
            #pragma unroll
            for (int i = 0; i < 8; ++i) { afrag[mi][i] = lo[i]; afrag[mi][i + 8] = hi[i]; }
        }
        v16h bfrag[4];
        #pragma unroll
        for (int ni = 0; ni < 4; ++ni) {
            const _Float16* p = &Bs[cur][(waveN * 64 + ni * 16 + lm) * LDSS];
            v8h lo = *(const v8h*)(p + r0);
            v8h hi = *(const v8h*)(p + r0 + 16);
            #pragma unroll
            for (int i = 0; i < 8; ++i) { bfrag[ni][i] = lo[i]; bfrag[ni][i + 8] = hi[i]; }
        }
        #pragma unroll
        for (int mi = 0; mi < 2; ++mi)
            #pragma unroll
            for (int ni = 0; ni < 4; ++ni)
                acc[mi][ni] = __builtin_amdgcn_wmma_f32_16x16x32_f16(
                    false, afrag[mi], false, bfrag[ni], (short)0, acc[mi][ni], false, false);

        if (kt + 1 < ktiles) stageToLds(cur ^ 1);
        __syncthreads();
    }

    const int rowBase = by * BM + waveM * 32 + ((lane >> 4) << 3);
    const int colBase = bx * BN + waveN * 64 + lm;
    #pragma unroll
    for (int mi = 0; mi < 2; ++mi)
        #pragma unroll
        for (int ni = 0; ni < 4; ++ni)
            #pragma unroll
            for (int r = 0; r < 8; ++r)
                C[(size_t)(rowBase + mi * 16 + r) * N + colBase + ni * 16] = acc[mi][ni][r];
}

extern "C" void kernel_launch(void* const* d_in, const int* in_sizes, int n_in,
                              void* d_out, int out_size, void* d_ws, size_t ws_size,
                              hipStream_t stream) {
    const float* A = (const float*)d_in[0];   // inputs [M, K] f32
    const float* B = (const float*)d_in[1];   // kernel [K, N] f32
    float* C = (float*)d_out;                 // [M, N] f32

    const int K = 4096;
    const int N = in_sizes[1] / K;   // 4096
    const int M = in_sizes[0] / K;   // 8192

    const size_t needA = (size_t)M * K * sizeof(_Float16);
    const size_t needB = (size_t)N * K * sizeof(_Float16);

    if (ws_size >= needA + needB) {
        _Float16* Ah = (_Float16*)d_ws;
        _Float16* Bt = (_Float16*)((char*)d_ws + needA);

        const size_t nA = (size_t)M * K;
        prep_a_f16<<<dim3((unsigned)(nA / (256 * 8))), 256, 0, stream>>>(A, Ah, nA);
        prep_b_tern_t<<<dim3(N / TB, K / TB), 256, 0, stream>>>(B, Bt, K, N);
        gemm_f16_wmma<<<dim3(N / GBN, M / GBM), 256, 0, stream>>>(Ah, Bt, C, M, N, K);
    } else {
        ternary_gemm_fused<<<dim3(N / BN, M / BM), 256, 0, stream>>>(A, B, C, M, N, K);
    }
}
